// MultiHeadAttention_77635828842510
// MI455X (gfx1250) — matrix operand order repaired and checked by reading, not compiled, not run on hardware
//
#include <hip/hip_runtime.h>
#include <hip/hip_bf16.h>

// ---------------------------------------------------------------------------
// MultiHeadAttention forward for MI455X (gfx1250, wave32, WMMA).
// Pipeline: convert->bf16, QKV GEMM (WMMA bf16), flash attention (WMMA bf16
// for QK^T and PV, exp2-domain online softmax), output projection GEMM.
// Straight-copy tiles staged by the Tensor Data Mover (TENSOR_LOAD_TO_LDS,
// one DMA per tile, TENSORcnt); transposed tiles register-staged.
// ---------------------------------------------------------------------------

#define D_MODEL 1024
#define N_HEADS 16
#define D_HEAD  64
#define SEQ     2048
#define BATCH   2
#define M_ROWS  (BATCH * SEQ)      // 4096

typedef __attribute__((ext_vector_type(16))) __bf16 v16bf;
typedef __attribute__((ext_vector_type(8)))  __bf16 v8bf;
typedef __attribute__((ext_vector_type(4)))  __bf16 v4bf;
typedef __attribute__((ext_vector_type(8)))  float  v8f;
typedef __attribute__((ext_vector_type(4)))  unsigned vu4;
typedef __attribute__((ext_vector_type(8)))  int      vi8;
typedef __attribute__((ext_vector_type(4)))  int      vi4;

// mask applied BEFORE scaling in the reference: -1e9, then /8; we also fold
// log2(e) so the softmax runs in exp2 domain (native v_exp_f32).
#define SCL2    0.1803368801111204f      // 0.125 * log2(e)
#define MASKED2 (-1.803368801e8f)        // -1e9 * SCL2

// ---------------------------------------------------------------------------
// Tensor Data Mover: build D# group1 (ISA 8.4) for a 2D bf16 tile copy with
// LDS row padding.  data_size=1 (2B). pad_interval v: pad after 2^(v+1) DWORDs;
// pad_amount v: insert (v+1) DWORDs.
// ---------------------------------------------------------------------------
__device__ __forceinline__ vi8 tdm_make_g1(unsigned tensor_d0, unsigned tensor_d1,
                                           unsigned tile_d0, unsigned tile_d1,
                                           unsigned long long stride0,
                                           unsigned pad_interval, unsigned pad_amount) {
  vi8 g;
  g[0] = (int)((1u << 16) | (1u << 20) | (pad_interval << 22) | (pad_amount << 25));
  g[1] = (int)((tensor_d0 & 0xFFFFu) << 16);                 // abar=0; td0[15:0]
  g[2] = (int)((tensor_d0 >> 16) | ((tensor_d1 & 0xFFFFu) << 16));
  g[3] = (int)((tensor_d1 >> 16) | (tile_d0 << 16));
  g[4] = (int)(tile_d1 & 0xFFFFu);                           // tile_dim2 = 0
  g[5] = (int)(unsigned)(stride0 & 0xFFFFFFFFu);             // stride0[31:0]
  g[6] = (int)(unsigned)((stride0 >> 32) & 0xFFFFu);         // stride0[47:32]
  g[7] = 0;
  return g;
}

// One TDM DMA: global tile (described by g1) -> LDS at lds_addr.  Issue from a
// single wave; completion via TENSORcnt (only the issuing wave waits).
__device__ __forceinline__ void tdm_load_2d(const void* gsrc, unsigned lds_addr, vi8 g1) {
  unsigned long long ga = (unsigned long long)(size_t)gsrc;
  vu4 g0;
  g0[0] = 1u;                                   // count=1, user mode
  g0[1] = lds_addr;                             // D# lds_addr
  g0[2] = (unsigned)ga;                         // global_addr[31:0]
  g0[3] = (unsigned)(ga >> 32) | (2u << 30);    // global_addr[56:32] | type=2
  vi4 z4 = (vi4)0;
#if defined(__clang_major__) && (__clang_major__ >= 23)
  vi8 z8 = (vi8)0;
  __builtin_amdgcn_tensor_load_to_lds(g0, g1, z4, z4, z8, 0);
#else
  __builtin_amdgcn_tensor_load_to_lds(g0, g1, z4, z4, 0);
#endif
}
__device__ __forceinline__ void tdm_wait0() {
  __builtin_amdgcn_s_wait_tensorcnt(0);
}

// ---------------------------------------------------------------------------
// Fragment loaders, matching CDNA5 ISA 7.12.2 VGPR layouts (wave32).
// A (16x32 bf16): lane L holds row m=L&15, koff=(L>>4)*8; K = koff+0..7 and
//   koff+16..23 -> two 16-byte loads.
// B (32x16 bf16): lane L holds col n=L&15, kb=(L>>4)*16; K = kb/2+0..7 and
//   kb/2+16..23 (same order as A) -> two 16-byte reads from an LDS tile [n][k].
// ---------------------------------------------------------------------------
__device__ __forceinline__ v16bf load_afrag(const __bf16* row, int koff) {
  v8bf lo = *reinterpret_cast<const v8bf*>(row + koff);
  v8bf hi = *reinterpret_cast<const v8bf*>(row + koff + 16);
  v16bf a;
#pragma unroll
  for (int e = 0; e < 8; ++e) { a[e] = lo[e]; a[e + 8] = hi[e]; }
  return a;
}

__device__ __forceinline__ v16bf load_bfrag(const __bf16* colrow, int kb) {
  v8bf lo = *reinterpret_cast<const v8bf*>(colrow + (kb & ~31) + ((kb & 16) >> 1));
  v8bf hi = *reinterpret_cast<const v8bf*>(colrow + (kb & ~31) + ((kb & 16) >> 1) + 16);
  v16bf b;
#pragma unroll
  for (int e = 0; e < 8; ++e) { b[e] = lo[e]; b[e + 8] = hi[e]; }
  return b;
}

// ---------------------------------------------------------------------------
// fp32 -> bf16 convert (vectorized x4)
// ---------------------------------------------------------------------------
__global__ __launch_bounds__(256) void cvt_f32_to_bf16(
    const float* __restrict__ in, __bf16* __restrict__ out, int n4) {
  int i = blockIdx.x * 256 + threadIdx.x;
  if (i < n4) {
    float4 f = reinterpret_cast<const float4*>(in)[i];
    v4bf o;
    o[0] = (__bf16)f.x; o[1] = (__bf16)f.y; o[2] = (__bf16)f.z; o[3] = (__bf16)f.w;
    reinterpret_cast<v4bf*>(out)[i] = o;
  }
}

// ---------------------------------------------------------------------------
// Tiled bf16 WMMA GEMM:  C[M,N] = A[M,K] @ B[K,N] + bias[N]
// Block tile 128x128, 8 waves (4xM x 2xN), wave tile 32x64, K-step 32.
// Double buffered: A tile via one TDM DMA per K-step (wave 0 issues/waits),
// B tile register-staged with transpose into LDS [n][k].
// ---------------------------------------------------------------------------
template <bool OUT_BF16>
__global__ __launch_bounds__(256) void gemm_bf16_wmma(
    const __bf16* __restrict__ A, const __bf16* __restrict__ B,
    const float* __restrict__ bias, void* __restrict__ Cout,
    int M, int N, int K) {
  constexpr int BM = 128, BN = 128, BK = 32;
  constexpr int LDA = 40, LDB = 40;          // 80B rows = 64B data + 16B pad
  __shared__ __bf16 lA[2][BM * LDA];
  __shared__ __bf16 lBt[2][BN * LDB];

  const int tid  = threadIdx.x;
  const int lane = tid & 31;
  const int wid  = tid >> 5;
  const int wm   = wid & 3;
  const int wn   = wid >> 2;
  const int ln   = lane & 15;
  const int half = lane >> 4;
  const int m0   = blockIdx.y * BM;
  const int n0   = blockIdx.x * BN;

  v8f acc[2][4];
#pragma unroll
  for (int mf = 0; mf < 2; ++mf)
#pragma unroll
    for (int nf = 0; nf < 4; ++nf)
#pragma unroll
      for (int i = 0; i < 8; ++i) acc[mf][nf][i] = 0.0f;

  const int brow = tid >> 3, bc0 = (tid & 7) * 16;   // B: 32 rows x 128 cols

  const __bf16* gAblk = A + (size_t)m0 * K;          // tile rows start here
  const __bf16* gB    = B + (size_t)brow * N + n0 + bc0;

  // D# group1 for the A tile: 128 rows x 32 bf16, row stride K elems,
  // LDS pad 16B after every 64B (pad_interval=3 -> 16 DW, pad_amount=3 -> 4 DW)
  const vi8 g1A = tdm_make_g1((unsigned)K, (unsigned)M, BK, BM,
                              (unsigned long long)K, 3u, 3u);

  // ---- prologue: stage kt=0 into buffer 0 ----
  if (wid == 0)
    tdm_load_2d(gAblk, (unsigned)(size_t)&lA[0][0], g1A);
  {
    v8bf b0 = *reinterpret_cast<const v8bf*>(gB);
    v8bf b1 = *reinterpret_cast<const v8bf*>(gB + 8);
#pragma unroll
    for (int e = 0; e < 8; ++e) {
      lBt[0][(bc0 + e) * LDB + brow]     = b0[e];
      lBt[0][(bc0 + 8 + e) * LDB + brow] = b1[e];
    }
  }
  if (wid == 0) tdm_wait0();
  __syncthreads();

  for (int kt = 0; kt < K; kt += BK) {
    const int cur = (kt >> 5) & 1;
    const int nxt = cur ^ 1;
    const bool has_next = (kt + BK) < K;

    v8bf b0, b1;
    if (has_next) {  // issue next tile early: TDM A -> LDS, B -> regs
      if (wid == 0)
        tdm_load_2d(gAblk + kt + BK, (unsigned)(size_t)&lA[nxt][0], g1A);
      const __bf16* gBn = gB + (size_t)(kt + BK) * N;
      b0 = *reinterpret_cast<const v8bf*>(gBn);
      b1 = *reinterpret_cast<const v8bf*>(gBn + 8);
      if (kt + 2 * BK < K)
        __builtin_prefetch(gB + (size_t)(kt + 2 * BK) * N, 0, 0);
    }

    // ---- compute current tile ----
    const int koff = half * 8;
    const int kb   = half * 16;
    v16bf af[2], bfv[4];
#pragma unroll
    for (int mf = 0; mf < 2; ++mf)
      af[mf] = load_afrag(&lA[cur][(wm * 32 + mf * 16 + ln) * LDA], koff);
#pragma unroll
    for (int nf = 0; nf < 4; ++nf)
      bfv[nf] = load_bfrag(&lBt[cur][(wn * 64 + nf * 16 + ln) * LDB], kb);
#pragma unroll
    for (int mf = 0; mf < 2; ++mf)
#pragma unroll
      for (int nf = 0; nf < 4; ++nf)
        acc[mf][nf] = __builtin_amdgcn_wmma_f32_16x16x32_bf16(
            false, af[mf], false, bfv[nf], (short)0, acc[mf][nf], false, false);

    if (has_next) {  // commit staged B, ensure TDM A arrived, flip buffers
#pragma unroll
      for (int e = 0; e < 8; ++e) {
        lBt[nxt][(bc0 + e) * LDB + brow]     = b0[e];
        lBt[nxt][(bc0 + 8 + e) * LDB + brow] = b1[e];
      }
      if (wid == 0) tdm_wait0();
      __syncthreads();
    }
  }

  // ---- epilogue: bias add, store (C layout: n = lane&15, m = i + 8*half) ----
#pragma unroll
  for (int mf = 0; mf < 2; ++mf)
#pragma unroll
    for (int nf = 0; nf < 4; ++nf) {
      const int n = n0 + wn * 64 + nf * 16 + ln;
      const float bv = bias[n];
#pragma unroll
      for (int i = 0; i < 8; ++i) {
        const int m = m0 + wm * 32 + mf * 16 + i + half * 8;
        const float v = acc[mf][nf][i] + bv;
        if (OUT_BF16)
          reinterpret_cast<__bf16*>(Cout)[(size_t)m * N + n] = (__bf16)v;
        else
          reinterpret_cast<float*>(Cout)[(size_t)m * N + n] = v;
      }
    }
}

// ---------------------------------------------------------------------------
// Flash attention, causal, mask-before-scale (reference semantics), exp2
// domain. Block = 256 q-rows of one (batch, head); 8 waves x 32 rows each.
// K/V tiles of 64 keys double-buffered: K via one TDM DMA per tile, V
// register-staged transposed. K fragments hoisted once per tile (shared by
// both strips); V fragments loaded per strip to stay inside the low VGPR bank.
// ---------------------------------------------------------------------------
__global__ __launch_bounds__(256) void attn_flash_wmma(
    const __bf16* __restrict__ qkv, __bf16* __restrict__ ctx) {
  constexpr int LDK = 72, LDV = 72, LDP = 72;          // 144B = 128B data + 16B pad
  __shared__ __bf16 lK[2][64 * LDK];                   // K tile  [j][d]
  __shared__ __bf16 lVt[2][64 * LDV];                  // V tile  [d][j] (transposed)
  __shared__ __bf16 lP[8 * 16 * LDP];                  // per-wave P pane [16][64]

  const int tid  = threadIdx.x;
  const int lane = tid & 31;
  const int wid  = tid >> 5;
  const int ln   = lane & 15;
  const int half = lane >> 4;
  const int qb   = blockIdx.x * 256;
  const int h    = blockIdx.y;
  const int b    = blockIdx.z;

  const size_t rs = 3 * D_MODEL;                       // qkv row stride (elems)
  const __bf16* qbase = qkv + (size_t)b * SEQ * rs + h * D_HEAD;
  const __bf16* kbase = qbase + D_MODEL;
  const __bf16* vbase = qbase + 2 * D_MODEL;

  // D# group1 for a K tile: 64 rows x 64 bf16, row stride 3072 elems,
  // LDS pad 16B after every 128B (pad_interval=4 -> 32 DW, pad_amount=3 -> 4 DW)
  const vi8 g1K = tdm_make_g1(D_HEAD, SEQ, 64u, 64u,
                              (unsigned long long)rs, 4u, 3u);

  // Q fragments stay in registers: wave rows qb + wid*32 + {0..31}
  v16bf qf[2][2];
#pragma unroll
  for (int mf = 0; mf < 2; ++mf)
#pragma unroll
    for (int kd = 0; kd < 2; ++kd) {
      const __bf16* qp = qbase + (size_t)(qb + wid * 32 + mf * 16 + ln) * rs;
      qf[mf][kd] = load_afrag(qp, kd * 32 + half * 8);
    }

  v8f oacc[2][4];
#pragma unroll
  for (int mf = 0; mf < 2; ++mf)
#pragma unroll
    for (int nf = 0; nf < 4; ++nf)
#pragma unroll
      for (int i = 0; i < 8; ++i) oacc[mf][nf][i] = 0.0f;

  float rowm[16], rowl[16];
#pragma unroll
  for (int r = 0; r < 16; ++r) { rowm[r] = -3.0e38f; rowl[r] = 0.0f; }

  const int jl = tid >> 2, c0 = (tid & 3) * 16;        // V cooperative load map
  const int jend = qb + 256;                           // causal: only j < qb+256
  __bf16* pane = &lP[wid * 16 * LDP];

  // ---- prologue: stage tile j0=0 into buffer 0 ----
  if (wid == 0)
    tdm_load_2d(kbase, (unsigned)(size_t)&lK[0][0], g1K);
  {
    const __bf16* gv = vbase + (size_t)jl * rs + c0;
    v8bf vv0 = *reinterpret_cast<const v8bf*>(gv);
    v8bf vv1 = *reinterpret_cast<const v8bf*>(gv + 8);
#pragma unroll
    for (int e = 0; e < 8; ++e) {
      lVt[0][(c0 + e) * LDV + jl]     = vv0[e];
      lVt[0][(c0 + 8 + e) * LDV + jl] = vv1[e];
    }
  }
  if (wid == 0) tdm_wait0();
  __syncthreads();

  for (int j0 = 0; j0 < jend; j0 += 64) {
    const int cur = (j0 >> 6) & 1;
    const int nxt = cur ^ 1;
    const bool has_next = (j0 + 64) < jend;

    v8bf vv0, vv1;
    if (has_next) {  // issue next K tile DMA, pull next V tile into regs
      if (wid == 0)
        tdm_load_2d(kbase + (size_t)(j0 + 64) * rs,
                    (unsigned)(size_t)&lK[nxt][0], g1K);
      const __bf16* gv = vbase + (size_t)(j0 + 64 + jl) * rs + c0;
      vv0 = *reinterpret_cast<const v8bf*>(gv);
      vv1 = *reinterpret_cast<const v8bf*>(gv + 8);
      if (j0 + 128 < jend)
        __builtin_prefetch(vbase + (size_t)(j0 + 128 + jl) * rs + c0, 0, 0);
    }

    // hoist K fragments once per tile; shared by both row strips
    v16bf kfr[2][4];
#pragma unroll
    for (int kd = 0; kd < 2; ++kd)
#pragma unroll
      for (int jf = 0; jf < 4; ++jf)
        kfr[kd][jf] = load_bfrag(&lK[cur][(jf * 16 + ln) * LDK], kd * 32 + half * 16);

    const bool need_mask = (j0 >= qb);  // masking only possible in last 4 tiles

#pragma unroll
    for (int mf = 0; mf < 2; ++mf) {
      // S = Q @ K^T for this 16-row strip (4 j-frags x 2 k-steps)
      v8f sf[4];
#pragma unroll
      for (int jf = 0; jf < 4; ++jf) {
        v8f s;
#pragma unroll
        for (int i = 0; i < 8; ++i) s[i] = 0.0f;
#pragma unroll
        for (int kd = 0; kd < 2; ++kd)
          s = __builtin_amdgcn_wmma_f32_16x16x32_bf16(
              false, qf[mf][kd], false, kfr[kd][jf], (short)0, s, false, false);
        sf[jf] = s;
      }

      // scale into exp2 domain (mask first, per reference), then tile row-max
      const int qrow0 = qb + wid * 32 + mf * 16;
      float tmax[8];
#pragma unroll
      for (int i = 0; i < 8; ++i) tmax[i] = -3.0e38f;
      if (need_mask) {
#pragma unroll
        for (int jf = 0; jf < 4; ++jf) {
          const int kidx = j0 + jf * 16 + ln;
#pragma unroll
          for (int i = 0; i < 8; ++i) {
            const int q = qrow0 + i + half * 8;
            float s = sf[jf][i] * SCL2;
            s = (kidx <= q) ? s : MASKED2;
            sf[jf][i] = s;
            tmax[i] = fmaxf(tmax[i], s);
          }
        }
      } else {
#pragma unroll
        for (int jf = 0; jf < 4; ++jf)
#pragma unroll
          for (int i = 0; i < 8; ++i) {
            const float s = sf[jf][i] * SCL2;
            sf[jf][i] = s;
            tmax[i] = fmaxf(tmax[i], s);
          }
      }
#pragma unroll
      for (int i = 0; i < 8; ++i) {  // reduce over the 16 n-lanes (half-wave)
        float t = tmax[i];
        t = fmaxf(t, __shfl_xor(t, 1));
        t = fmaxf(t, __shfl_xor(t, 2));
        t = fmaxf(t, __shfl_xor(t, 4));
        t = fmaxf(t, __shfl_xor(t, 8));
        tmax[i] = t;
      }

      float mn[8], alpha[8], psum[8];
#pragma unroll
      for (int i = 0; i < 8; ++i) {
        const float mo = rowm[mf * 8 + i];
        mn[i] = fmaxf(mo, tmax[i]);
        alpha[i] = __builtin_amdgcn_exp2f(mo - mn[i]);
        rowm[mf * 8 + i] = mn[i];
        psum[i] = 0.0f;
      }
#pragma unroll
      for (int jf = 0; jf < 4; ++jf)
#pragma unroll
        for (int i = 0; i < 8; ++i) {
          const float p = __builtin_amdgcn_exp2f(sf[jf][i] - mn[i]);
          sf[jf][i] = p;
          psum[i] += p;
        }
#pragma unroll
      for (int i = 0; i < 8; ++i) {
        float t = psum[i];
        t += __shfl_xor(t, 1);
        t += __shfl_xor(t, 2);
        t += __shfl_xor(t, 4);
        t += __shfl_xor(t, 8);
        rowl[mf * 8 + i] = rowl[mf * 8 + i] * alpha[i] + t;
      }

      // rescale running context; spill P (bf16) to the wave-private pane
#pragma unroll
      for (int nf = 0; nf < 4; ++nf)
#pragma unroll
        for (int i = 0; i < 8; ++i) oacc[mf][nf][i] *= alpha[i];

#pragma unroll
      for (int jf = 0; jf < 4; ++jf)
#pragma unroll
        for (int i = 0; i < 8; ++i)
          pane[(i + half * 8) * LDP + jf * 16 + ln] = (__bf16)sf[jf][i];

      // O[mf] += P @ V  (P re-read in A-frag layout; V frags per k-step batch)
#pragma unroll
      for (int kd = 0; kd < 2; ++kd) {
        v16bf pa = load_afrag(&pane[ln * LDP], kd * 32 + half * 8);
        v16bf vfr[4];
#pragma unroll
        for (int nf = 0; nf < 4; ++nf)
          vfr[nf] = load_bfrag(&lVt[cur][(nf * 16 + ln) * LDV], kd * 32 + half * 16);
#pragma unroll
        for (int nf = 0; nf < 4; ++nf)
          oacc[mf][nf] = __builtin_amdgcn_wmma_f32_16x16x32_bf16(
              false, pa, false, vfr[nf], (short)0, oacc[mf][nf], false, false);
      }
    }

    if (has_next) {  // commit staged V, ensure TDM K arrived, flip buffers
#pragma unroll
      for (int e = 0; e < 8; ++e) {
        lVt[nxt][(c0 + e) * LDV + jl]     = vv0[e];
        lVt[nxt][(c0 + 8 + e) * LDV + jl] = vv1[e];
      }
      if (wid == 0) tdm_wait0();
      __syncthreads();
    }
  }

  // normalize and write ctx [B,S,D] bf16 (heads interleaved at h*64)
  __bf16* cb = ctx + (size_t)b * SEQ * D_MODEL + h * D_HEAD;
#pragma unroll
  for (int mf = 0; mf < 2; ++mf)
#pragma unroll
    for (int nf = 0; nf < 4; ++nf) {
      const int d = nf * 16 + ln;
#pragma unroll
      for (int i = 0; i < 8; ++i) {
        const int srow = qb + wid * 32 + mf * 16 + i + half * 8;
        const float v = oacc[mf][nf][i] / rowl[mf * 8 + i];
        cb[(size_t)srow * D_MODEL + d] = (__bf16)v;
      }
    }
}

// ---------------------------------------------------------------------------
// Launch
// ---------------------------------------------------------------------------
extern "C" void kernel_launch(void* const* d_in, const int* in_sizes, int n_in,
                              void* d_out, int out_size, void* d_ws, size_t ws_size,
                              hipStream_t stream) {
  const float* x     = (const float*)d_in[0];   // [2,2048,1024]
  const float* Wqkv  = (const float*)d_in[1];   // [1024,3072]
  const float* bqkv  = (const float*)d_in[2];   // [3072]
  const float* Wproj = (const float*)d_in[3];   // [1024,1024]
  const float* bproj = (const float*)d_in[4];   // [1024]
  float* out = (float*)d_out;                   // [2,2048,1024]

  char* ws = (char*)d_ws;
  __bf16* xb     = (__bf16*)(ws);                          //  8,388,608 B
  __bf16* Wqkvb  = (__bf16*)(ws + 8388608);                //  6,291,456 B
  __bf16* Wprojb = (__bf16*)(ws + 14680064);               //  2,097,152 B
  __bf16* qkvb   = (__bf16*)(ws + 16777216);               // 25,165,824 B
  __bf16* ctxb   = (__bf16*)(ws + 41943040);               //  8,388,608 B  (total ~50.3 MB)

  // 1) fp32 -> bf16 converts
  cvt_f32_to_bf16<<<(M_ROWS * D_MODEL / 4 + 255) / 256, 256, 0, stream>>>(x, xb, M_ROWS * D_MODEL / 4);
  cvt_f32_to_bf16<<<(D_MODEL * 3 * D_MODEL / 4 + 255) / 256, 256, 0, stream>>>(Wqkv, Wqkvb, D_MODEL * 3 * D_MODEL / 4);
  cvt_f32_to_bf16<<<(D_MODEL * D_MODEL / 4 + 255) / 256, 256, 0, stream>>>(Wproj, Wprojb, D_MODEL * D_MODEL / 4);

  // 2) QKV projection: [4096,1024] @ [1024,3072] + b -> bf16
  gemm_bf16_wmma<true><<<dim3(3 * D_MODEL / 128, M_ROWS / 128), 256, 0, stream>>>(
      xb, Wqkvb, bqkv, (void*)qkvb, M_ROWS, 3 * D_MODEL, D_MODEL);

  // 3) causal flash attention per (qtile, head, batch)
  attn_flash_wmma<<<dim3(SEQ / 256, N_HEADS, BATCH), 256, 0, stream>>>(qkvb, ctxb);

  // 4) output projection: [4096,1024] @ [1024,1024] + b -> fp32 d_out
  gemm_bf16_wmma<false><<<dim3(D_MODEL / 128, M_ROWS / 128), 256, 0, stream>>>(
      ctxb, Wprojb, bproj, (void*)out, M_ROWS, D_MODEL, D_MODEL);
}
